// AttentionModel_71975061946633
// MI455X (gfx1250) — matrix operand order repaired and checked by reading, not compiled, not run on hardware
//
#include <hip/hip_runtime.h>
#include <hip/hip_bf16.h>

// Problem constants (from reference)
#define BB 64
#define NN 2000
#define NNP 2016         // padded node count for transposed gv rows (tail tile unguarded)
#define DD 128
#define HH 8
#define BEAM 16
#define EXPAND 16
#define DH 16
#define NT ((NN) / 16)   // 125 node tiles of 16

typedef __attribute__((ext_vector_type(16))) __bf16 v16bf;
typedef __attribute__((ext_vector_type(8)))  float  f32x8;
typedef unsigned int u32x4 __attribute__((ext_vector_type(4)));
typedef int          i32x8 __attribute__((ext_vector_type(8)));
typedef int          i32x4 __attribute__((ext_vector_type(4)));

#define NEGF (-3.4028235e38f)

#if defined(__has_builtin)
#if __has_builtin(__builtin_amdgcn_tensor_load_to_lds) && \
    __has_builtin(__builtin_amdgcn_s_wait_tensorcnt)
#define USE_TDM 1
#endif
#endif

// ---- WMMA fragment index maps (CDNA5 ISA 7.12.2, wave32) -------------------
// A-matrix (16-bit, 16x32): lane r=lane&15 holds row M=r; element j ->
//   K = 16*laneGroup + j (contiguous k order chosen here, not the ISA table order)
__device__ __forceinline__ int kmapA(int j, int lg) {
    return 16 * lg + j;
}
// B-matrix (16-bit, 32x16): lane r = column N; element j -> K = 16*lg + j (same contiguous order as A, chosen here)
// C/D (f32 16x16): element i -> row M = i + 8*lg, column N = r

__device__ __forceinline__ f32x8 zero8() {
    f32x8 z;
#pragma unroll
    for (int i = 0; i < 8; ++i) z[i] = 0.f;
    return z;
}

__device__ __forceinline__ float redmax16(float x) {
#pragma unroll
    for (int off = 1; off < 16; off <<= 1) x = fmaxf(x, __shfl_xor(x, off, 16));
    return x;
}
__device__ __forceinline__ float redsum16(float x) {
#pragma unroll
    for (int off = 1; off < 16; off <<= 1) x += __shfl_xor(x, off, 16);
    return x;
}

// ---- K0: weight transpose + bf16 convert -----------------------------------
__global__ __launch_bounds__(256) void k_conv_wnode(const float* __restrict__ Wn,
                                                    __bf16* __restrict__ Wtn) {
    int i = blockIdx.x * 256 + threadIdx.x;           // over 384*128
    if (i < 384 * 128) {
        int c = i / 128, k = i % 128;
        Wtn[i] = (__bf16)Wn[k * 384 + c];             // Wtn[c][k]
    }
}
__global__ __launch_bounds__(256) void k_conv_wout(const float* __restrict__ Wo,
                                                   __bf16* __restrict__ Wto) {
    int i = blockIdx.x * 256 + threadIdx.x;           // over 128*128
    if (i < 128 * 128) {
        int c = i / 128, k = i % 128;
        Wto[i] = (__bf16)Wo[k * 128 + c];             // Wto[c][k]
    }
}

// ---- mean over nodes -------------------------------------------------------
__global__ __launch_bounds__(128) void k_mean(const float* __restrict__ emb,
                                              float* __restrict__ meanb) {
    int b = blockIdx.x, d = threadIdx.x;
    float s = 0.f;
    for (int n = 0; n < NN; ++n) s += emb[((size_t)b * NN + n) * DD + d];
    meanb[b * DD + d] = s * (1.0f / NN);
}

// ---- query = mean@W_fixed + concat(e_first,e_last)@W_step ------------------
__global__ __launch_bounds__(128) void k_query(const float* __restrict__ emb,
                                               const float* __restrict__ meanb,
                                               const float* __restrict__ Wf,
                                               const float* __restrict__ Ws,
                                               const int* __restrict__ seq,
                                               float* __restrict__ query) {
    int bm = blockIdx.x, b = bm / BEAM, d = threadIdx.x;
    int s0 = seq[bm * 2 + 0], s1 = seq[bm * 2 + 1];
    const float* e0 = emb + ((size_t)b * NN + s0) * DD;
    const float* e1 = emb + ((size_t)b * NN + s1) * DD;
    const float* mb = meanb + b * DD;
    float acc = 0.f;
    for (int k = 0; k < DD; ++k) acc += mb[k] * Wf[k * DD + d];
    for (int k = 0; k < DD; ++k) acc += e0[k] * Ws[k * DD + d];
    for (int k = 0; k < DD; ++k) acc += e1[k] * Ws[(DD + k) * DD + d];
    query[bm * DD + d] = acc;
}

// ---- K1: kvl = emb @ W_node (bf16 WMMA, f32 accum) -> gk/gvT/lk ------------
// Block = (b, 16-node tile). TDM stages the 16x128 f32 tile into LDS with a
// 1-dword pad every 128 dwords (row stride 129 -> conflict-free A-frag reads).
// 4 waves, each owns 6 of the 24 output col-tiles.
__global__ __launch_bounds__(128) void k1_kvl(const float* __restrict__ emb,
                                              const __bf16* __restrict__ Wtn,
                                              __bf16* __restrict__ gk,
                                              __bf16* __restrict__ gvT,
                                              __bf16* __restrict__ lk) {
    int blk = blockIdx.x, b = blk / NT, nt = blk % NT;
    int tid = threadIdx.x;
    __shared__ float AsF[16 * 129];                   // 16 rows x 128 dw (+1 pad)
    const float* tile = emb + ((size_t)b * NN + nt * 16) * DD;  // contiguous 8KB

#if USE_TDM
    if (tid < 32) {                                   // wave 0 drives the TDM
        unsigned lds_off = (unsigned)(size_t)&AsF[0];
        unsigned long long ga = (unsigned long long)(size_t)tile;
        u32x4 g0;
        g0[0] = 1u;                                   // count=1 (valid user D#)
        g0[1] = lds_off;                              // lds_addr
        g0[2] = (unsigned)(ga & 0xffffffffull);       // global_addr[31:0]
        g0[3] = (unsigned)((ga >> 32) & 0x01ffffffull) | (2u << 30); // addr[56:32] | type=2
        i32x8 g1;
        g1[0] = (int)((2u << 16)    // data_size = 4 bytes
                    | (1u << 20)    // pad_enable
                    | (6u << 22));  // pad_interval: 128 dwords; pad_amount=0 -> 1 dword
        g1[1] = (int)(2048u << 16); // tensor_dim0 = 2048 elements (bits 79:48)
        g1[2] = (int)(1u << 16);    // tensor_dim1 = 1 (bits 111:80)
        g1[3] = (int)(2048u << 16); // tile_dim0 = 2048 (bits 127:112)
        g1[4] = 1;                  // tile_dim1 = 1
        g1[5] = 2048;               // tensor_dim0_stride = 2048
        g1[6] = 0;
        g1[7] = 0;
        i32x4 gz4 = {0, 0, 0, 0};
        i32x8 gz8 = {0, 0, 0, 0, 0, 0, 0, 0};
        __builtin_amdgcn_tensor_load_to_lds(g0, g1, gz4, gz4, gz8, 0);
        __builtin_amdgcn_s_wait_tensorcnt(0);
    }
    __syncthreads();
#else
    for (int i = tid; i < 16 * 128; i += 128) {
        int node = i >> 7, k = i & 127;
        AsF[node * 129 + k] = tile[node * 128 + k];
    }
    __syncthreads();
#endif

    int lane = tid & 31, wave = tid >> 5;
    int r = lane & 15, lg = lane >> 4;
#pragma unroll 1
    for (int t = 0; t < 6; ++t) {
        int ct = wave * 6 + t;                        // 0..23 (384/16 col tiles)
        f32x8 acc = zero8();
#pragma unroll
        for (int kc = 0; kc < 4; ++kc) {
            v16bf a, w;
#pragma unroll
            for (int j = 0; j < 16; ++j)
                a[j] = (__bf16)AsF[r * 129 + kc * 32 + kmapA(j, lg)];
#pragma unroll
            for (int j = 0; j < 16; ++j)
                w[j] = Wtn[(ct * 16 + r) * 128 + kc * 32 + 16 * lg + j];
            acc = __builtin_amdgcn_wmma_f32_16x16x32_bf16(false, a, false, w,
                                                          (short)0, acc, false, false);
        }
        int part = ct >> 3;                           // 0:gk 1:gv 2:lk
        if (part == 1) {
            // transposed gv: [b][h][dh][node], node contiguous per lane
            int h = ct & 7, dh = r;
            size_t rowb = ((size_t)(b * HH + h) * DH + dh) * NNP + nt * 16 + 8 * lg;
#pragma unroll
            for (int i = 0; i < 8; ++i)
                gvT[rowb + i] = (__bf16)acc[i];
        } else {
            int chb = (ct & 7) * 16;
            __bf16* out = (part == 0) ? gk : lk;
#pragma unroll
            for (int i = 0; i < 8; ++i) {
                int node = nt * 16 + i + 8 * lg;
                out[((size_t)b * NN + node) * DD + chb + r] = (__bf16)acc[i];
            }
        }
    }
}

// ---- K2: flash glimpse attention per (b,h) ---------------------------------
__global__ __launch_bounds__(32) void k2_attn(const float* __restrict__ query,
                                              const __bf16* __restrict__ gk,
                                              const __bf16* __restrict__ gvT,
                                              const unsigned char* __restrict__ mask,
                                              float* __restrict__ heads) {
    int blk = blockIdx.x, b = blk >> 3, h = blk & 7;
    int lane = threadIdx.x, r = lane & 15, lg = lane >> 4;
    __shared__ float Pl[16 * 33];                     // P transpose staging

    // Q A-fragment: rows = beams, K = dh padded 16->32 with zeros
    v16bf qa;
#pragma unroll
    for (int j = 0; j < 16; ++j) {
        int K = kmapA(j, lg);
        qa[j] = (__bf16)((K < DH) ? query[((size_t)b * BEAM + r) * DD + h * DH + K] : 0.f);
    }
    f32x8 O = zero8();
    float rmax[8], lsum[8];
#pragma unroll
    for (int i = 0; i < 8; ++i) { rmax[i] = NEGF; lsum[i] = 0.f; }
    const float sc = 0.25f;                           // 1/sqrt(dh=16)
    const __bf16* gvrow = gvT + ((size_t)(b * HH + h) * DH + r) * NNP;

    for (int base = 0; base < NN; base += 32) {
        // prefetch next gk/gv chunk (streams through L2)
        __builtin_prefetch(gk + ((size_t)b * NN + base + 32) * DD + h * DH, 0, 1);
        __builtin_prefetch(gvrow + base + 32, 0, 1);

        int n0 = base + r, n1 = base + 16 + r;
        // gk^T B-fragments (K = dh, padded; lg==1 lanes are the zero pad)
        v16bf kb0, kb1;
#pragma unroll
        for (int j = 0; j < 16; ++j) {
            kb0[j] = (lg == 0 && n0 < NN) ? gk[((size_t)b * NN + n0) * DD + h * DH + j]
                                          : (__bf16)0.f;
            kb1[j] = (lg == 0 && n1 < NN) ? gk[((size_t)b * NN + n1) * DD + h * DH + j]
                                          : (__bf16)0.f;
        }
        f32x8 s0 = __builtin_amdgcn_wmma_f32_16x16x32_bf16(false, qa, false, kb0,
                                                           (short)0, zero8(), false, false);
        f32x8 s1 = __builtin_amdgcn_wmma_f32_16x16x32_bf16(false, qa, false, kb1,
                                                           (short)0, zero8(), false, false);
        float p0[8], p1[8];
#pragma unroll
        for (int i = 0; i < 8; ++i) {
            int m = i + 8 * lg;
            bool m0 = (n0 >= NN) || mask[((size_t)b * BEAM + m) * NN + n0];
            bool m1 = (n1 >= NN) || mask[((size_t)b * BEAM + m) * NN + n1];
            float v0 = m0 ? NEGF : s0[i] * sc;
            float v1 = m1 ? NEGF : s1[i] * sc;
            float tm = redmax16(fmaxf(v0, v1));       // row max across node cols
            float nm = fmaxf(rmax[i], tm);
            float c = __expf(rmax[i] - nm);
            float e0 = m0 ? 0.f : __expf(v0 - nm);
            float e1 = m1 ? 0.f : __expf(v1 - nm);
            lsum[i] = lsum[i] * c + redsum16(e0 + e1);
            rmax[i] = nm;
            O[i] *= c;
            p0[i] = e0; p1[i] = e1;
        }
        // transpose P (C-layout -> A-layout) through LDS
        __syncthreads();
#pragma unroll
        for (int i = 0; i < 8; ++i) {
            int m = i + 8 * lg;
            Pl[m * 33 + r]      = p0[i];
            Pl[m * 33 + 16 + r] = p1[i];
        }
        __syncthreads();
        v16bf pa, vb;
#pragma unroll
        for (int j = 0; j < 16; ++j)
            pa[j] = (__bf16)Pl[r * 33 + kmapA(j, lg)];
        // gv B-frag: K = node (contiguous per lane in gvT), col = dh = r.
        // Tail overreads land in the NNP pad; they multiply P entries that the
        // mask path already zeroed (poison bf16 is finite, so 0*x == 0).
#pragma unroll
        for (int j = 0; j < 16; ++j)
            vb[j] = gvrow[base + 16 * lg + j];
        O = __builtin_amdgcn_wmma_f32_16x16x32_bf16(false, pa, false, vb,
                                                    (short)0, O, false, false);
    }
#pragma unroll
    for (int i = 0; i < 8; ++i) {
        int m = i + 8 * lg;
        float l = lsum[i];
        heads[((size_t)b * BEAM + m) * DD + h * DH + r] = (l > 0.f) ? O[i] / l : 0.f;
    }
}

// ---- K3a: glimpse = heads @ W_out ------------------------------------------
__global__ __launch_bounds__(32) void k3a_glimpse(const float* __restrict__ heads,
                                                  const __bf16* __restrict__ Wto,
                                                  __bf16* __restrict__ glimpse) {
    int b = blockIdx.x, lane = threadIdx.x, r = lane & 15, lg = lane >> 4;
#pragma unroll 1
    for (int ct = 0; ct < 8; ++ct) {
        f32x8 acc = zero8();
#pragma unroll
        for (int kc = 0; kc < 4; ++kc) {
            v16bf a, w;
#pragma unroll
            for (int j = 0; j < 16; ++j)
                a[j] = (__bf16)heads[((size_t)b * BEAM + r) * DD + kc * 32 + kmapA(j, lg)];
#pragma unroll
            for (int j = 0; j < 16; ++j)
                w[j] = Wto[(ct * 16 + r) * 128 + kc * 32 + 16 * lg + j];
            acc = __builtin_amdgcn_wmma_f32_16x16x32_bf16(false, a, false, w,
                                                          (short)0, acc, false, false);
        }
#pragma unroll
        for (int i = 0; i < 8; ++i)
            glimpse[((size_t)b * BEAM + i + 8 * lg) * DD + ct * 16 + r] = (__bf16)acc[i];
    }
}

// ---- K3b: logits = glimpse . lk / sqrt(D), tanh clip, mask -----------------
__global__ __launch_bounds__(32) void k3b_logits(const __bf16* __restrict__ glimpse,
                                                 const __bf16* __restrict__ lk,
                                                 const unsigned char* __restrict__ mask,
                                                 float* __restrict__ logp) {
    int blk = blockIdx.x, b = blk / NT, nt = blk % NT;
    int lane = threadIdx.x, r = lane & 15, lg = lane >> 4;
    f32x8 acc = zero8();
#pragma unroll
    for (int kc = 0; kc < 4; ++kc) {
        v16bf a, w;
#pragma unroll
        for (int j = 0; j < 16; ++j)
            a[j] = glimpse[((size_t)b * BEAM + r) * DD + kc * 32 + kmapA(j, lg)];
#pragma unroll
        for (int j = 0; j < 16; ++j)
            w[j] = lk[((size_t)b * NN + nt * 16 + r) * DD + kc * 32 + 16 * lg + j];
        acc = __builtin_amdgcn_wmma_f32_16x16x32_bf16(false, a, false, w,
                                                      (short)0, acc, false, false);
    }
    const float inv = 0.08838834764831845f;           // 1/sqrt(128)
#pragma unroll
    for (int i = 0; i < 8; ++i) {
        int m = i + 8 * lg, node = nt * 16 + r;
        float v = 10.f * tanhf(acc[i] * inv);
        bool msk = mask[((size_t)b * BEAM + m) * NN + node] != 0;
        logp[((size_t)b * BEAM + m) * NN + node] = msk ? NEGF : v;
    }
}

// ---- K3c: in-place log_softmax over N + top-16 (one wave per row) ----------
__global__ __launch_bounds__(32) void k3c_softmax_topk(float* __restrict__ logp,
                                                       float* __restrict__ topv,
                                                       float* __restrict__ topi) {
    int row = blockIdx.x;                             // b*BEAM + m
    int lane = threadIdx.x;
    float* rp = logp + (size_t)row * NN;
    float v[63];
#pragma unroll
    for (int j = 0; j < 63; ++j) {
        int idx = j * 32 + lane;
        v[j] = (idx < NN) ? rp[idx] : NEGF;
    }
    float mx = NEGF;
#pragma unroll
    for (int j = 0; j < 63; ++j) mx = fmaxf(mx, v[j]);
#pragma unroll
    for (int off = 16; off; off >>= 1) mx = fmaxf(mx, __shfl_xor(mx, off, 32));
    float s = 0.f;
#pragma unroll
    for (int j = 0; j < 63; ++j) s += __expf(v[j] - mx);
#pragma unroll
    for (int off = 16; off; off >>= 1) s += __shfl_xor(s, off, 32);
    float logZ = mx + __logf(s);
#pragma unroll
    for (int j = 0; j < 63; ++j) {
        int idx = j * 32 + lane;
        v[j] -= logZ;
        if (idx < NN) rp[idx] = v[j];
    }
    // iterative argmax top-16 (tie -> lower index, matching stable top_k)
    for (int k = 0; k < EXPAND; ++k) {
        float bv = NEGF;
        int bi = 0x7fffffff;
#pragma unroll
        for (int j = 0; j < 63; ++j) {
            int idx = j * 32 + lane;
            if (idx < NN && v[j] > bv) { bv = v[j]; bi = idx; }
        }
#pragma unroll
        for (int off = 16; off; off >>= 1) {
            float ov = __shfl_xor(bv, off, 32);
            int   oi = __shfl_xor(bi, off, 32);
            if (ov > bv || (ov == bv && oi < bi)) { bv = ov; bi = oi; }
        }
        if (lane == 0) {
            topv[row * EXPAND + k] = bv;
            topi[row * EXPAND + k] = (float)bi;       // int output flattened as float
        }
#pragma unroll
        for (int j = 0; j < 63; ++j)
            if (j * 32 + lane == bi) v[j] = NEGF;
    }
}

// ---- host-side orchestration ----------------------------------------------
extern "C" void kernel_launch(void* const* d_in, const int* in_sizes, int n_in,
                              void* d_out, int out_size, void* d_ws, size_t ws_size,
                              hipStream_t stream) {
    (void)in_sizes; (void)n_in; (void)out_size; (void)ws_size;
    const float* emb  = (const float*)d_in[0];        // [B,N,D]
    const float* Wn   = (const float*)d_in[1];        // [D,3D]
    const float* Wf   = (const float*)d_in[2];        // [D,D]
    const float* Ws   = (const float*)d_in[3];        // [2D,D]
    const float* Wo   = (const float*)d_in[4];        // [D,D]
    const int*   seq  = (const int*)d_in[5];          // [B,BEAM,2]
    const unsigned char* mask = (const unsigned char*)d_in[6]; // bool [B,BEAM,N]

    float* out_logp = (float*)d_out;                          // [B,BEAM,N]
    float* out_topv = out_logp + (size_t)BB * BEAM * NN;      // [B,BEAM,EXPAND]
    float* out_topi = out_topv + (size_t)BB * BEAM * EXPAND;  // [B,BEAM,EXPAND]

    // workspace carve (~100 MB)
    char* w = (char*)d_ws;
    auto carve = [&](size_t bytes) { void* p = (void*)w; w += (bytes + 255) & ~(size_t)255; return p; };
    __bf16* gk   = (__bf16*)carve((size_t)BB * NN * DD * 2);
    __bf16* gvT  = (__bf16*)carve((size_t)BB * HH * DH * NNP * 2);
    __bf16* lk   = (__bf16*)carve((size_t)BB * NN * DD * 2);
    __bf16* Wtn  = (__bf16*)carve((size_t)384 * 128 * 2);
    __bf16* Wto  = (__bf16*)carve((size_t)128 * 128 * 2);
    float*  meanb   = (float*)carve((size_t)BB * DD * 4);
    float*  query   = (float*)carve((size_t)BB * BEAM * DD * 4);
    float*  heads   = (float*)carve((size_t)BB * BEAM * DD * 4);
    __bf16* glimpse = (__bf16*)carve((size_t)BB * BEAM * DD * 2);

    k_conv_wnode<<<(384 * 128 + 255) / 256, 256, 0, stream>>>(Wn, Wtn);
    k_conv_wout<<<(128 * 128 + 255) / 256, 256, 0, stream>>>(Wo, Wto);
    k_mean<<<BB, 128, 0, stream>>>(emb, meanb);
    k_query<<<BB * BEAM, 128, 0, stream>>>(emb, meanb, Wf, Ws, seq, query);
    k1_kvl<<<BB * NT, 128, 0, stream>>>(emb, Wtn, gk, gvT, lk);
    k2_attn<<<BB * HH, 32, 0, stream>>>(query, gk, gvT, mask, heads);
    k3a_glimpse<<<BB, 32, 0, stream>>>(heads, Wto, glimpse);
    k3b_logits<<<BB * NT, 32, 0, stream>>>(glimpse, lk, mask, out_logp);
    k3c_softmax_topk<<<BB * BEAM, 32, 0, stream>>>(out_logp, out_topv, out_topi);
}
